// SpatiotemporalAttention_20504173871850
// MI455X (gfx1250) — matrix operand order repaired and checked by reading, not compiled, not run on hardware
//
#include <hip/hip_runtime.h>

typedef __bf16 bf16;
typedef __attribute__((ext_vector_type(16))) __bf16 v16bf;
typedef __attribute__((ext_vector_type(8)))  float  v8f;
typedef __attribute__((ext_vector_type(4)))  unsigned int v4u;
typedef __attribute__((ext_vector_type(4)))  int v4i;

#define BATCH  16
#define SEQ    2048
#define DIM    1024
#define HEADS  16
#define CLIPL  8
#define NCHUNK 256
#define DKH    64
#define NTOK   (BATCH*SEQ)

#if __has_builtin(__builtin_amdgcn_global_load_async_to_lds_b128)
#define HAS_ASYNC_LDS 1
#else
#define HAS_ASYNC_LDS 0
#endif

union FragA { v16bf v; v4u q[2]; };
union Pack8 { v4u q; bf16 h[8]; };

__device__ inline v8f v8f_zero() {
  v8f r;
#pragma unroll
  for (int i = 0; i < 8; ++i) r[i] = 0.0f;
  return r;
}

// 16-byte global -> LDS copy; async CDNA5 path (ASYNCcnt), sync fallback.
__device__ __forceinline__ void async_copy16(const bf16* g, bf16* l) {
#if HAS_ASYNC_LDS
  typedef __attribute__((address_space(1))) v4i GAvec;  // global (AS1) v4i
  typedef __attribute__((address_space(3))) v4i LAvec;  // LDS    (AS3) v4i
  __builtin_amdgcn_global_load_async_to_lds_b128(
      (GAvec*)(v4i*)(bf16*)g, (LAvec*)(v4i*)l, 0, 0);
#else
  *reinterpret_cast<v4u*>(l) = *reinterpret_cast<const v4u*>(g);
#endif
}

template <int N>
__device__ __forceinline__ void wait_async() {
#if HAS_ASYNC_LDS
#if __has_builtin(__builtin_amdgcn_s_wait_asynccnt)
  __builtin_amdgcn_s_wait_asynccnt(N);
#else
  asm volatile("s_wait_asynccnt %0" ::"n"(N) : "memory");
#endif
#endif
}

// -------------------------------------------------------------------------
// f32 -> bf16 conversion
// -------------------------------------------------------------------------
__global__ void cvt_f32_bf16(const float* __restrict__ in, bf16* __restrict__ out, int n) {
  int i = blockIdx.x * 256 + threadIdx.x;
  if (i < n) out[i] = (bf16)in[i];
}

// -------------------------------------------------------------------------
// f32 [K][N] -> bf16 transposed [N][K] (so GEMM B-tiles are contiguous and
// async-copyable). LDS-tiled 32x32 so both sides stay coalesced.
// -------------------------------------------------------------------------
__global__ __launch_bounds__(256) void cvt_transpose(
    const float* __restrict__ in, bf16* __restrict__ out, int K, int N) {
  __shared__ float tile[32][33];
  const int kb = blockIdx.y * 32, nb = blockIdx.x * 32;
  const int tx = threadIdx.x & 31, ty = threadIdx.x >> 5;  // ty: 0..7
  for (int i = ty; i < 32; i += 8)
    tile[i][tx] = in[(size_t)(kb + i) * N + nb + tx];
  __syncthreads();
  for (int i = ty; i < 32; i += 8)
    out[(size_t)(nb + i) * K + kb + tx] = (bf16)tile[tx][i];
}

// -------------------------------------------------------------------------
// Tiled WMMA GEMM: C[M,N] = A[M,K] @ Wt[N,K]^T (+ A1 @ Wt1^T if DUAL) + bias
// BM=128 BN=128 BK=32, 256 threads (8 waves), wave -> 32x64 patch.
// Double-buffered LDS, async global->LDS copies (4 per thread per tile).
// -------------------------------------------------------------------------
template <bool DUAL, bool STOREF>
__global__ __launch_bounds__(256) void gemm_wmma(
    const bf16* __restrict__ A0, const bf16* __restrict__ Wt0,
    const bf16* __restrict__ A1, const bf16* __restrict__ Wt1,
    const float* __restrict__ bias,
    bf16* __restrict__ outB, float* __restrict__ outF,
    int M, int N, int K) {
  __shared__ __align__(16) bf16 As[2][128 * 32];  // [m][k]
  __shared__ __align__(16) bf16 Bs[2][128 * 32];  // [n][k]

  const int tid  = threadIdx.x;
  const int lane = tid & 31, wave = tid >> 5;
  const int wm = wave & 3, wn = wave >> 2;
  const int m0 = blockIdx.y * 128, n0 = blockIdx.x * 128;
  const int grp = lane >> 4, lm = lane & 15;
  const int KT = K / 32;                 // k-tiles per pass
  const int T = (DUAL ? 2 : 1) * KT;     // total tiles

  v8f acc[2][4];
#pragma unroll
  for (int i = 0; i < 2; ++i)
#pragma unroll
    for (int j = 0; j < 4; ++j) acc[i][j] = v8f_zero();

  auto issue = [&](int t, int buf) {
    const int pass = DUAL ? (t >= KT) : 0;
    const int k0 = (t - pass * KT) * 32;
    const bf16* A  = pass ? A1  : A0;
    const bf16* Wt = pass ? Wt1 : Wt0;
#pragma unroll
    for (int i = 0; i < 2; ++i) {
      int s = tid * 2 + i;
      int row = s >> 2, c = (s & 3) * 8;
      async_copy16(A  + (size_t)(m0 + row) * K + k0 + c, &As[buf][row * 32 + c]);
      async_copy16(Wt + (size_t)(n0 + row) * K + k0 + c, &Bs[buf][row * 32 + c]);
    }
  };

  issue(0, 0);
  for (int t = 0; t < T; ++t) {
    const int cur = t & 1;
    if (t + 1 < T) {
      issue(t + 1, cur ^ 1);
      wait_async<4>();   // retire current tile's 4 asyncs (in-order)
    } else {
      wait_async<0>();
    }
    __syncthreads();

    FragA a[2], b[4];
#pragma unroll
    for (int mi = 0; mi < 2; ++mi) {
      const bf16* base = &As[cur][(wm * 32 + mi * 16 + lm) * 32];
      a[mi].q[0] = *reinterpret_cast<const v4u*>(base + grp * 8);
      a[mi].q[1] = *reinterpret_cast<const v4u*>(base + 16 + grp * 8);
    }
#pragma unroll
    for (int ni = 0; ni < 4; ++ni) {
      const bf16* base = &Bs[cur][(wn * 64 + ni * 16 + lm) * 32];
      b[ni].q[0] = *reinterpret_cast<const v4u*>(base + grp * 8);
      b[ni].q[1] = *reinterpret_cast<const v4u*>(base + 16 + grp * 8);
    }
#pragma unroll
    for (int mi = 0; mi < 2; ++mi)
#pragma unroll
      for (int ni = 0; ni < 4; ++ni)
        acc[mi][ni] = __builtin_amdgcn_wmma_f32_16x16x32_bf16(
            false, a[mi].v, false, b[ni].v, (short)0, acc[mi][ni], false, false);
    __syncthreads();  // protect buf cur before it is refilled next iter
  }

  // epilogue: C layout lane->(n=lm, m=r+8*grp)
#pragma unroll
  for (int mi = 0; mi < 2; ++mi)
#pragma unroll
    for (int ni = 0; ni < 4; ++ni) {
      int col = n0 + wn * 64 + ni * 16 + lm;
      float bv = bias ? bias[col] : 0.0f;
#pragma unroll
      for (int r = 0; r < 8; ++r) {
        int row = m0 + wm * 32 + mi * 16 + r + grp * 8;
        float val = acc[mi][ni][r] + bv;
        if (STOREF) outF[(size_t)row * N + col] = val;
        else        outB[(size_t)row * N + col] = (bf16)val;
      }
    }
}

// -------------------------------------------------------------------------
// Spatial attention: L=8 per sequence, VALU path. 1 block = (b, nc) chunk.
// -------------------------------------------------------------------------
__global__ __launch_bounds__(128) void spatial_attn(
    const bf16* __restrict__ Q, const bf16* __restrict__ Km,
    const bf16* __restrict__ V, const int* __restrict__ mask,
    bf16* __restrict__ out) {
  __shared__ __align__(16) bf16 Ks[CLIPL * DIM];
  __shared__ __align__(16) bf16 Vs[CLIPL * DIM];
  const int bid = blockIdx.x;
  const int b = bid / NCHUNK, nc = bid % NCHUNK;
  const size_t t0 = (size_t)b * SEQ + (size_t)nc * CLIPL;
  const int tid = threadIdx.x;

  for (int c8 = tid; c8 < CLIPL * DIM / 8; c8 += 128) {
    async_copy16(Km + t0 * DIM + c8 * 8, Ks + c8 * 8);
    async_copy16(V  + t0 * DIM + c8 * 8, Vs + c8 * 8);
  }
  wait_async<0>();
  __syncthreads();

  const int h = tid >> 3, c = tid & 7;
  const int cb = h * DKH;
  float qr[DKH];
#pragma unroll
  for (int d = 0; d < DKH; ++d) qr[d] = (float)Q[(t0 + c) * DIM + cb + d];

  float sc[CLIPL];
  float mx = -1e30f;
#pragma unroll
  for (int kc = 0; kc < CLIPL; ++kc) {
    float acc = 0.0f;
#pragma unroll
    for (int d = 0; d < DKH; ++d) acc += qr[d] * (float)Ks[kc * DIM + cb + d];
    acc *= 0.125f;  // 1/sqrt(64)
    if (mask[b * SEQ + nc * CLIPL + kc] == 0) acc = -1e30f;
    sc[kc] = acc;
    mx = fmaxf(mx, acc);
  }
  float sum = 0.0f;
#pragma unroll
  for (int kc = 0; kc < CLIPL; ++kc) { sc[kc] = __expf(sc[kc] - mx); sum += sc[kc]; }
  float inv = 1.0f / sum;
#pragma unroll
  for (int d = 0; d < DKH; ++d) {
    float acc = 0.0f;
#pragma unroll
    for (int kc = 0; kc < CLIPL; ++kc) acc += sc[kc] * (float)Vs[kc * DIM + cb + d];
    out[(t0 + c) * DIM + cb + d] = (bf16)(acc * inv);
  }
}

// -------------------------------------------------------------------------
// Temporal attention: flash-attention over L=256, dk=64 via WMMA.
// Block = (qtile, head, b*CLIP); 8 waves, wave owns 16 query rows.
// -------------------------------------------------------------------------
__global__ __launch_bounds__(256) void temporal_attn(
    const bf16* __restrict__ Q, const bf16* __restrict__ Km,
    const bf16* __restrict__ V, const int* __restrict__ mask,
    bf16* __restrict__ out) {
  __shared__ __align__(16) bf16 Qs[128 * 64];
  __shared__ __align__(16) bf16 Ks[32 * 64];     // [key][d]
  __shared__ __align__(16) bf16 Vs[64 * 32];     // [d][key] transposed
  __shared__ __align__(16) bf16 Ps[8 * 16 * 32]; // per-wave P tiles

  const int qt = blockIdx.x;   // 0..1 (query tile of 128)
  const int h = blockIdx.y;    // head
  const int b = blockIdx.z / CLIPL;
  const int clip = blockIdx.z % CLIPL;
  const int tid = threadIdx.x, lane = tid & 31, wave = tid >> 5;
  const int grp = lane >> 4, lm = lane & 15;
  const size_t colbase = (size_t)h * DKH;

  // load Q tile (async, 4 b128 chunks per thread)
  for (int c = tid; c < 128 * 8; c += 256) {
    int r = c >> 3, c8 = (c & 7) * 8;
    int nc = qt * 128 + r;
    size_t tok = (size_t)b * SEQ + (size_t)nc * CLIPL + clip;
    async_copy16(Q + tok * DIM + colbase + c8, Qs + r * 64 + c8);
  }
  wait_async<0>();
  __syncthreads();

  // per-wave Q A-fragments (16 rows x 64 dk -> two 16x32 frags)
  FragA qa[2];
  {
    const bf16* base = Qs + (wave * 16 + lm) * 64;
#pragma unroll
    for (int ks = 0; ks < 2; ++ks) {
      qa[ks].q[0] = *reinterpret_cast<const v4u*>(base + ks * 32 + grp * 8);
      qa[ks].q[1] = *reinterpret_cast<const v4u*>(base + ks * 32 + 16 + grp * 8);
    }
  }

  float mrow[8], lrow[8];
  v8f o[4];
#pragma unroll
  for (int r = 0; r < 8; ++r) { mrow[r] = -1e30f; lrow[r] = 0.0f; }
#pragma unroll
  for (int ni = 0; ni < 4; ++ni) o[ni] = v8f_zero();

  for (int j = 0; j < NCHUNK / 32; ++j) {
    __syncthreads();
    // K block 32x64: async contiguous rows
    {
      int kr = tid >> 3, c8 = (tid & 7) * 8;  // 256 threads = 256 chunks
      int nc = j * 32 + kr;
      size_t tok = (size_t)b * SEQ + (size_t)nc * CLIPL + clip;
      async_copy16(Km + tok * DIM + colbase + c8, Ks + kr * 64 + c8);
    }
    // V block transposed through VGPRs: 1 b128 read + 8 b16 scatters/thread
    {
      int kr = tid >> 3, d8 = (tid & 7) * 8;
      int nc = j * 32 + kr;
      size_t tok = (size_t)b * SEQ + (size_t)nc * CLIPL + clip;
      Pack8 u;
      u.q = *reinterpret_cast<const v4u*>(V + tok * DIM + colbase + d8);
#pragma unroll
      for (int e = 0; e < 8; ++e) Vs[(d8 + e) * 32 + kr] = u.h[e];
    }
    wait_async<0>();
    __syncthreads();

    // scores: Q(16x64) @ K^T -> two 16x16 tiles over 32 keys
    v8f s[2];
#pragma unroll
    for (int kn = 0; kn < 2; ++kn) {
      s[kn] = v8f_zero();
#pragma unroll
      for (int ks = 0; ks < 2; ++ks) {
        FragA kb;
        const bf16* base = Ks + (kn * 16 + lm) * 64 + ks * 32 + grp * 8;
        kb.q[0] = *reinterpret_cast<const v4u*>(base);
        kb.q[1] = *reinterpret_cast<const v4u*>(base + 16);
        s[kn] = __builtin_amdgcn_wmma_f32_16x16x32_bf16(
            false, qa[ks].v, false, kb.v, (short)0, s[kn], false, false);
      }
    }

    // online softmax over this 32-key block
    int key0 = j * 32 + lm;
    bool ok0 = mask[b * SEQ + key0 * CLIPL + clip] != 0;
    bool ok1 = mask[b * SEQ + (key0 + 16) * CLIPL + clip] != 0;
#pragma unroll
    for (int r = 0; r < 8; ++r) {
      float v0 = ok0 ? s[0][r] * 0.125f : -1e30f;
      float v1 = ok1 ? s[1][r] * 0.125f : -1e30f;
      float t = fmaxf(v0, v1);
#pragma unroll
      for (int off = 1; off < 16; off <<= 1) t = fmaxf(t, __shfl_xor(t, off, 32));
      float mn = fmaxf(mrow[r], t);
      float scale = __expf(mrow[r] - mn);
      float p0 = __expf(v0 - mn);
      float p1 = __expf(v1 - mn);
      float ps = p0 + p1;
#pragma unroll
      for (int off = 1; off < 16; off <<= 1) ps += __shfl_xor(ps, off, 32);
      lrow[r] = lrow[r] * scale + ps;
      mrow[r] = mn;
#pragma unroll
      for (int ni = 0; ni < 4; ++ni) o[ni][r] *= scale;
      int prow = r + grp * 8;
      Ps[wave * 512 + prow * 32 + lm] = (bf16)p0;
      Ps[wave * 512 + prow * 32 + lm + 16] = (bf16)p1;
    }

    // P(16x32) as A-frag (wave-private LDS region, DS ops in-order per wave)
    FragA pa;
    {
      const bf16* base = Ps + wave * 512 + lm * 32;
      pa.q[0] = *reinterpret_cast<const v4u*>(base + grp * 8);
      pa.q[1] = *reinterpret_cast<const v4u*>(base + 16 + grp * 8);
    }
    // P @ V -> accumulate 16x64 output
#pragma unroll
    for (int ni = 0; ni < 4; ++ni) {
      FragA vb;
      const bf16* base = Vs + (ni * 16 + lm) * 32 + grp * 8;
      vb.q[0] = *reinterpret_cast<const v4u*>(base);
      vb.q[1] = *reinterpret_cast<const v4u*>(base + 16);
      o[ni] = __builtin_amdgcn_wmma_f32_16x16x32_bf16(
          false, pa.v, false, vb.v, (short)0, o[ni], false, false);
    }
  }

  // final normalize + store (token order already un-transposed)
#pragma unroll
  for (int ni = 0; ni < 4; ++ni)
#pragma unroll
    for (int r = 0; r < 8; ++r) {
      int nc = qt * 128 + wave * 16 + r + grp * 8;
      size_t tok = (size_t)b * SEQ + (size_t)nc * CLIPL + clip;
      float val = o[ni][r] / lrow[r];
      out[tok * DIM + colbase + ni * 16 + lm] = (bf16)val;
    }
}

// -------------------------------------------------------------------------
// Host orchestration
// -------------------------------------------------------------------------
extern "C" void kernel_launch(void* const* d_in, const int* in_sizes, int n_in,
                              void* d_out, int out_size, void* d_ws, size_t ws_size,
                              hipStream_t stream) {
  (void)in_sizes; (void)n_in; (void)out_size; (void)ws_size;
  const float* x = (const float*)d_in[0];
  const int* mask = (const int*)d_in[1];
  // weights order: sa_q, sa_k, sa_v, sa_o, ta_q, ta_k, ta_v, ta_o, fuse
  const float* wf[9];
  const float* bfp[9];
  for (int i = 0; i < 8; ++i) {
    wf[i] = (const float*)d_in[2 + 2 * i];
    bfp[i] = (const float*)d_in[3 + 2 * i];
  }
  wf[8] = (const float*)d_in[18];
  bfp[8] = (const float*)d_in[19];

  char* base = (char*)d_ws;
  size_t off = 0;
  auto carve = [&](size_t elems) -> bf16* {
    bf16* r = (bf16*)(base + off);
    off += (elems * sizeof(bf16) + 255) & ~(size_t)255;
    return r;
  };
  bf16* xb = carve((size_t)NTOK * DIM);
  bf16* wtb[10];  // transposed bf16 weights [N][K]; 8 proj + 2 fuse halves
  for (int i = 0; i < 10; ++i) wtb[i] = carve((size_t)DIM * DIM);
  bf16* qs = carve((size_t)NTOK * DIM);
  bf16* ks = carve((size_t)NTOK * DIM);
  bf16* vs = carve((size_t)NTOK * DIM);
  bf16* qt = carve((size_t)NTOK * DIM);
  bf16* kt = carve((size_t)NTOK * DIM);
  bf16* vt = carve((size_t)NTOK * DIM);
  bf16* attn_s = carve((size_t)NTOK * DIM);
  bf16* attn_t = carve((size_t)NTOK * DIM);
  bf16* proj_s = qs;  // q buffers dead after attention -> reuse
  bf16* proj_t = qt;

  // 1) precision conversion (+ weight transpose so GEMM tiles are contiguous)
  int nx = NTOK * DIM;
  cvt_f32_bf16<<<(nx + 255) / 256, 256, 0, stream>>>(x, xb, nx);
  dim3 tgrid(DIM / 32, DIM / 32);
  for (int i = 0; i < 8; ++i)
    cvt_transpose<<<tgrid, 256, 0, stream>>>(wf[i], wtb[i], DIM, DIM);
  cvt_transpose<<<tgrid, 256, 0, stream>>>(wf[8], wtb[8], DIM, DIM);                     // fuse rows 0..D-1
  cvt_transpose<<<tgrid, 256, 0, stream>>>(wf[8] + (size_t)DIM * DIM, wtb[9], DIM, DIM); // fuse rows D..2D-1

  dim3 ggrid(DIM / 128, NTOK / 128);
  // 2) QKV projections (spatial + temporal share input x)
  gemm_wmma<false, false><<<ggrid, 256, 0, stream>>>(xb, wtb[0], nullptr, nullptr, bfp[0], qs, nullptr, NTOK, DIM, DIM);
  gemm_wmma<false, false><<<ggrid, 256, 0, stream>>>(xb, wtb[1], nullptr, nullptr, bfp[1], ks, nullptr, NTOK, DIM, DIM);
  gemm_wmma<false, false><<<ggrid, 256, 0, stream>>>(xb, wtb[2], nullptr, nullptr, bfp[2], vs, nullptr, NTOK, DIM, DIM);
  gemm_wmma<false, false><<<ggrid, 256, 0, stream>>>(xb, wtb[4], nullptr, nullptr, bfp[4], qt, nullptr, NTOK, DIM, DIM);
  gemm_wmma<false, false><<<ggrid, 256, 0, stream>>>(xb, wtb[5], nullptr, nullptr, bfp[5], kt, nullptr, NTOK, DIM, DIM);
  gemm_wmma<false, false><<<ggrid, 256, 0, stream>>>(xb, wtb[6], nullptr, nullptr, bfp[6], vt, nullptr, NTOK, DIM, DIM);

  // 3) attention
  spatial_attn<<<BATCH * NCHUNK, 128, 0, stream>>>(qs, ks, vs, mask, attn_s);
  temporal_attn<<<dim3(2, HEADS, BATCH * CLIPL), 256, 0, stream>>>(qt, kt, vt, mask, attn_t);

  // 4) output projections
  gemm_wmma<false, false><<<ggrid, 256, 0, stream>>>(attn_s, wtb[3], nullptr, nullptr, bfp[3], proj_s, nullptr, NTOK, DIM, DIM);
  gemm_wmma<false, false><<<ggrid, 256, 0, stream>>>(attn_t, wtb[7], nullptr, nullptr, bfp[7], proj_t, nullptr, NTOK, DIM, DIM);

  // 5) fused concat-matmul: out = proj_s @ fuse_w[:D] + proj_t @ fuse_w[D:] + b
  gemm_wmma<true, true><<<ggrid, 256, 0, stream>>>(
      proj_s, wtb[8], proj_t, wtb[9], bfp[8],
      nullptr, (float*)d_out, NTOK, DIM, DIM);
}
